// DeepSet_TM_36404142800957
// MI455X (gfx1250) — compile-verified, matrix operand order checked by reading, not run on hardware
//
#include <hip/hip_runtime.h>

typedef __bf16 bf16;
typedef __attribute__((ext_vector_type(16))) __bf16 v16bf;
typedef __attribute__((ext_vector_type(8)))  __bf16 v8bf;
typedef __attribute__((ext_vector_type(8)))  float  v8f;
typedef int v4i __attribute__((__vector_size__(16)));

#define B_SZ 64
#define N_SZ 1024
#define DIN  512
#define DH   1024
#define NOUT 10

#define HAS_ASYNC_LDS __has_builtin(__builtin_amdgcn_global_load_async_to_lds_b128)

typedef __attribute__((address_space(1))) v4i* gv4i_p;
typedef __attribute__((address_space(3))) v4i* lv4i_p;

__device__ __forceinline__ bf16 to_bf16(float f) {
  unsigned u = __builtin_bit_cast(unsigned, f);
  u += 0x7FFFu + ((u >> 16) & 1u);              // round-to-nearest-even
  unsigned short h = (unsigned short)(u >> 16);
  return __builtin_bit_cast(bf16, h);
}

__device__ __forceinline__ void copy16(v8bf* lds_dst, const bf16* gsrc) {
#if HAS_ASYNC_LDS
  __builtin_amdgcn_global_load_async_to_lds_b128(
      (gv4i_p)gsrc, (lv4i_p)lds_dst, 0, 0);
#else
  *lds_dst = *(const v8bf*)gsrc;
#endif
}
__device__ __forceinline__ void async_wait0() {
#if HAS_ASYNC_LDS
  asm volatile("s_wait_asynccnt 0x0" ::: "memory");
#endif
}
__device__ __forceinline__ void async_wait8() {
#if HAS_ASYNC_LDS
  asm volatile("s_wait_asynccnt 0x8" ::: "memory");
#endif
}

// ---------------------------------------------------------------- converts
__global__ void cvt_f32_bf16(const float* __restrict__ in, bf16* __restrict__ out, int n) {
  int i = blockIdx.x * blockDim.x + threadIdx.x;
  if (i < n) out[i] = to_bf16(in[i]);
}

// W fp32 [K][Nc] -> Wt bf16 [Nc][K]
__global__ void cvt_transpose_bf16(const float* __restrict__ W, bf16* __restrict__ Wt,
                                   int K, int Nc) {
  int i = blockIdx.x * blockDim.x + threadIdx.x;
  if (i < K * Nc) {
    int k = i / Nc, n = i - k * Nc;
    Wt[(size_t)n * K + k] = to_bf16(W[i]);
  }
}

// ---------------------------------------------------------------- WMMA GEMM
// C[M,Nc] = act(A[M,K] @ W[K,Nc] + bias).  A bf16 row-major, W pre-transposed
// bf16 [Nc][K].  Tile: 128(M) x 128(N) x 64(K), double-buffered async stages.
// Fragment-major LDS: chunk q (8 bf16): c=q&1, L=(q>>1)&31, ks=(q>>6)&1, t=q>>7
//   A: m = t*16+(L&15), k = ks*32 + ((L>>4)+2c)*8      (t = tm, 0..7)
//   B: n = t*16+(L&15), k = ks*32 + ((L>>4)+2c)*8      (t = tn, 0..7)
__device__ __forceinline__ void stage_copy(v8bf* As, v8bf* Bs,
                                           const bf16* A, const bf16* Wt,
                                           int row0, int col0, int k0,
                                           int M, int K, int tid) {
#pragma unroll
  for (int t = 0; t < 4; ++t) {            // A: 1024 chunks
    int q  = tid + 256 * t;
    int c  = q & 1;
    int L  = (q >> 1) & 31;
    int ks = (q >> 6) & 1;
    int tm = q >> 7;
    int m  = tm * 16 + (L & 15);
    int kst = ks * 32 + ((L >> 4) + 2 * c) * 8;
    int gr = row0 + m; if (gr > M - 1) gr = M - 1;   // clamp, discarded at write
    copy16(&As[q], A + (size_t)gr * K + (size_t)(k0 + kst));
  }
#pragma unroll
  for (int t = 0; t < 4; ++t) {            // B: 1024 chunks
    int q  = tid + 256 * t;
    int c  = q & 1;
    int L  = (q >> 1) & 31;
    int ks = (q >> 6) & 1;
    int tn = q >> 7;
    int n  = tn * 16 + (L & 15);
    int kst = ks * 32 + ((L >> 4) + 2 * c) * 8;
    copy16(&Bs[q], Wt + (size_t)(col0 + n) * K + (size_t)(k0 + kst));
  }
}

__device__ __forceinline__ v16bf frag_ld(const v8bf* base, int idx) {
  v8bf lo = base[idx], hi = base[idx + 1];
  return __builtin_shufflevector(lo, hi, 0,1,2,3,4,5,6,7,8,9,10,11,12,13,14,15);
}

template <bool RELU, bool OUTBF16>
__global__ void __launch_bounds__(256)
gemm_wmma(const bf16* __restrict__ A, const bf16* __restrict__ Wt,
          const float* __restrict__ bias, float* __restrict__ Cf,
          bf16* __restrict__ Cb, int M, int K, int Nc) {
  __shared__ v8bf As[2][1024];   // 2 x 16 KB
  __shared__ v8bf Bs[2][1024];   // 2 x 16 KB

  const int tid  = threadIdx.x;
  const int lane = tid & 31;
  const int wid  = tid >> 5;          // wave owns subtile row tm = wid
  const int row0 = blockIdx.y * 128;
  const int col0 = blockIdx.x * 128;
  const int mrow = lane & 15;
  const int khalf = lane >> 4;

  v8f acc[8];
#pragma unroll
  for (int i = 0; i < 8; ++i) acc[i] = {};

  const int nst = K / 64;
  stage_copy(As[0], Bs[0], A, Wt, row0, col0, 0, M, K, tid);

  for (int s = 0; s < nst; ++s) {
    const int cur = s & 1;
    if (s + 1 < nst) {
      stage_copy(As[(s + 1) & 1], Bs[(s + 1) & 1], A, Wt,
                 row0, col0, (s + 1) * 64, M, K, tid);
      async_wait8();                 // newest 8 = next stage; current has landed
    } else {
      async_wait0();
    }
    __syncthreads();

    // software-pipelined fragment consumption: load frag tn+1 before WMMA tn
#pragma unroll
    for (int ks = 0; ks < 2; ++ks) {
      v16bf a    = frag_ld(As[cur], ((wid * 2 + ks) * 32 + lane) * 2);
      v16bf bcur = frag_ld(Bs[cur], ((0 * 2 + ks) * 32 + lane) * 2);
#pragma unroll
      for (int tn = 0; tn < 8; ++tn) {
        v16bf bnext = bcur;
        if (tn < 7)
          bnext = frag_ld(Bs[cur], (((tn + 1) * 2 + ks) * 32 + lane) * 2);
        acc[tn] = __builtin_amdgcn_wmma_f32_16x16x32_bf16(
            false, a, false, bcur, (short)0, acc[tn], false, false);
        bcur = bnext;
      }
    }
    __syncthreads();
  }

  // epilogue: VGPR r -> M=r (lanes 0-15) / r+8 (lanes 16-31), N = lane&15
#pragma unroll
  for (int tn = 0; tn < 8; ++tn) {
#pragma unroll
    for (int r = 0; r < 8; ++r) {
      int m = wid * 16 + r + khalf * 8;
      int n = tn * 16 + mrow;
      int gr = row0 + m, gc = col0 + n;
      if (gr < M) {
        float v = acc[tn][r] + bias[gc];
        if (RELU) v = fmaxf(v, 0.0f);
        if (OUTBF16) Cb[(size_t)gr * Nc + gc] = to_bf16(v);
        else         Cf[(size_t)gr * Nc + gc] = v;
      }
    }
  }
}

// ---------------------------------------------------------------- trimmed mean
__global__ void __launch_bounds__(256)
trim_mean(const float* __restrict__ enc, const float* __restrict__ mask,
          float* __restrict__ agg) {
  __shared__ float sv[8][N_SZ];     // 32 KB
  __shared__ float red[256];
  __shared__ int s_nv;
  const int tid = threadIdx.x;
  const int b   = blockIdx.x >> 7;            // DH/8 = 128 column tiles
  const int h0  = (blockIdx.x & 127) * 8;

  float pm = 0.f;
  for (int n = tid; n < N_SZ; n += 256) pm += mask[b * N_SZ + n];
  red[tid] = pm;
  __syncthreads();
  for (int off = 128; off > 0; off >>= 1) {
    if (tid < off) red[tid] += red[tid + off];
    __syncthreads();
  }
  if (tid == 0) s_nv = (int)red[0];
  __syncthreads();
  const int nv = s_nv;
  const int kk = (int)((float)nv * 0.1f);     // trunc, matches astype(int32)

  for (int i = tid; i < 8 * N_SZ; i += 256) {
    int n = i >> 3, c = i & 7;
    float m = mask[b * N_SZ + n];
    float e = enc[((size_t)(b * N_SZ + n)) * DH + (h0 + c)];
    sv[c][n] = (m > 0.f) ? e * m : __builtin_inff();
  }
  __syncthreads();

  for (int k = 2; k <= N_SZ; k <<= 1) {
    for (int j = k >> 1; j > 0; j >>= 1) {
      for (int t = tid; t < 8 * (N_SZ / 2); t += 256) {
        int col = t >> 9;
        int p   = t & 511;
        int i   = ((p & ~(j - 1)) << 1) | (p & (j - 1));
        int ij  = i | j;
        bool up = ((i & k) == 0);
        float a = sv[col][i], c2 = sv[col][ij];
        if ((a > c2) == up) { sv[col][i] = c2; sv[col][ij] = a; }
      }
      __syncthreads();
    }
  }

  const int col  = tid >> 5;
  const int lane = tid & 31;
  const int lo = kk, hi = nv - kk;
  float s = 0.f;
  for (int i = lane; i < N_SZ; i += 32)
    if (i >= lo && i < hi) s += sv[col][i];
  for (int off = 16; off > 0; off >>= 1) s += __shfl_down(s, off, 32);
  if (lane == 0) agg[b * DH + h0 + col] = s / (float)(nv - 2 * kk);
}

// ---------------------------------------------------------------- tiny head
__global__ void __launch_bounds__(256)
head10(const float* __restrict__ H3, const float* __restrict__ W4,
       const float* __restrict__ b4, float* __restrict__ out) {
  int gw   = (blockIdx.x * 256 + threadIdx.x) >> 5;
  int lane = threadIdx.x & 31;
  if (gw >= B_SZ * NOUT) return;
  int b = gw / NOUT, o = gw % NOUT;
  float s = 0.f;
  for (int h = lane; h < DH; h += 32)
    s += H3[b * DH + h] * W4[h * NOUT + o];
  for (int off = 16; off > 0; off >>= 1) s += __shfl_down(s, off, 32);
  if (lane == 0) out[b * NOUT + o] = s + b4[o];
}

// ---------------------------------------------------------------- launch
extern "C" void kernel_launch(void* const* d_in, const int* in_sizes, int n_in,
                              void* d_out, int out_size, void* d_ws, size_t ws_size,
                              hipStream_t stream) {
  (void)in_sizes; (void)n_in; (void)out_size; (void)ws_size;
  const float* X    = (const float*)d_in[0];
  const float* mask = (const float*)d_in[1];
  const float* W1   = (const float*)d_in[2];
  const float* b1   = (const float*)d_in[3];
  const float* W2   = (const float*)d_in[4];
  const float* b2   = (const float*)d_in[5];
  const float* W3   = (const float*)d_in[6];
  const float* b3   = (const float*)d_in[7];
  const float* W4   = (const float*)d_in[8];
  const float* b4   = (const float*)d_in[9];
  float* out = (float*)d_out;

  char* ws = (char*)d_ws;
  size_t off = 0;
  bf16* Xb  = (bf16*)(ws + off); off += (size_t)B_SZ * N_SZ * DIN * 2;  // 67 MB
  bf16* W1t = (bf16*)(ws + off); off += (size_t)DIN * DH * 2;           // [DH][DIN]
  bf16* W2t = (bf16*)(ws + off); off += (size_t)DH * DH * 2;            // [DH][DH]
  bf16* W3t = (bf16*)(ws + off); off += (size_t)DH * DH * 2;
  bf16* H1b = (bf16*)(ws + off); off += (size_t)B_SZ * N_SZ * DH * 2;   // 134 MB
  float* ENC = (float*)(ws + off); off += (size_t)B_SZ * N_SZ * DH * 4; // 268 MB
  float* AGG = (float*)(ws + off); off += (size_t)B_SZ * DH * 4;
  bf16*  AGGb = (bf16*)(ws + off); off += (size_t)B_SZ * DH * 2;
  float* H3  = (float*)(ws + off); off += (size_t)B_SZ * DH * 4;

  const int nX = B_SZ * N_SZ * DIN;
  cvt_f32_bf16<<<(nX + 255) / 256, 256, 0, stream>>>(X, Xb, nX);
  cvt_transpose_bf16<<<(DIN * DH + 255) / 256, 256, 0, stream>>>(W1, W1t, DIN, DH);
  cvt_transpose_bf16<<<(DH * DH + 255) / 256, 256, 0, stream>>>(W2, W2t, DH, DH);
  cvt_transpose_bf16<<<(DH * DH + 255) / 256, 256, 0, stream>>>(W3, W3t, DH, DH);

  // H1 = relu(X @ W1 + b1) -> bf16
  dim3 g1(DH / 128, (B_SZ * N_SZ) / 128);
  gemm_wmma<true, true><<<g1, 256, 0, stream>>>(
      Xb, W1t, b1, nullptr, H1b, B_SZ * N_SZ, DIN, DH);
  // ENC = H1 @ W2 + b2 -> fp32
  gemm_wmma<false, false><<<g1, 256, 0, stream>>>(
      H1b, W2t, b2, ENC, nullptr, B_SZ * N_SZ, DH, DH);

  trim_mean<<<B_SZ * (DH / 8), 256, 0, stream>>>(ENC, mask, AGG);
  cvt_f32_bf16<<<(B_SZ * DH + 255) / 256, 256, 0, stream>>>(AGG, AGGb, B_SZ * DH);

  // H3 = relu(AGG @ W3 + b3) -> fp32  (M=64 < 128: clamped staging, guarded writes)
  dim3 g3(DH / 128, 1);
  gemm_wmma<true, false><<<g3, 256, 0, stream>>>(
      AGGb, W3t, b3, H3, nullptr, B_SZ, DH, DH);

  head10<<<(B_SZ * NOUT * 32 + 255) / 256, 256, 0, stream>>>(H3, W4, b4, out);
}
